// SelfAttention_687194767973
// MI455X (gfx1250) — matrix operand order repaired and checked by reading, not compiled, not run on hardware
//
#include <hip/hip_runtime.h>
#include <hip/hip_bf16.h>

typedef __attribute__((ext_vector_type(16))) _Float16 v16h;
typedef __attribute__((ext_vector_type(8)))  float    v8f;

union Frag16 { v16h h; unsigned int u[8]; };

#define L_SEQ 4096
#define D_MODEL 512
#define N_HEAD 8
#define DH 64
#define BATCH 2

// ---------------------------------------------------------------------------
// Projection: y = x @ W^T  (per-wave 16x16 tile, K-loop of 32, f32->f16 WMMA)
// MODE 0: Q -> [BH, L, 64], scaled by dh^-0.5
// MODE 1: K -> d-pair interleaved [BH, 32, L, 2]   (B-fragment layout for QK^T)
// MODE 2: V -> k-pair interleaved [BH, 2048, 64, 2] (B-fragment layout for P*V)
// ---------------------------------------------------------------------------
template <int MODE>
__global__ __launch_bounds__(128)
void proj_kernel(const float* __restrict__ X,   // [B*L, D]
                 const float* __restrict__ W,   // [D, D] row-major
                 _Float16* __restrict__ out)
{
    const int lane  = threadIdx.x & 31;
    const int wave  = threadIdx.x >> 5;
    const int laneM = lane & 15;
    const int hi    = (lane >> 4) & 1;
    const int m0    = blockIdx.x * 16;                    // row tile in [0, B*L)
    const int n0    = (blockIdx.y * 4 + wave) * 16;       // col tile in [0, D)

    v8f acc = {};
    const float* xrow = X + (size_t)(m0 + laneM) * D_MODEL;
    const float* wrow = W + (size_t)(n0 + laneM) * D_MODEL;

    for (int kc = 0; kc < D_MODEL; kc += 32) {
        Frag16 a, b;
        // A fragment: lanes 0-15 K=0..15, lanes 16-31 K=16..31 (contiguous)
        #pragma unroll
        for (int v = 0; v < 8; ++v) {
            int k = kc + hi * 16 + 2 * v;
            float2 xv = *(const float2*)(xrow + k);
            a.h[2 * v]     = (_Float16)xv.x;
            a.h[2 * v + 1] = (_Float16)xv.y;
        }
        // B fragment: B[k][n] = W[n][k]; lanes 0-15 rows K=0..15, lanes 16-31 K=16..31
        #pragma unroll
        for (int v = 0; v < 8; ++v) {
            float2 wv = *(const float2*)(wrow + kc + hi * 16 + 2 * v);
            b.h[2 * v]     = (_Float16)wv.x;
            b.h[2 * v + 1] = (_Float16)wv.y;
        }
        acc = __builtin_amdgcn_wmma_f32_16x16x32_f16(false, a.h, false, b.h,
                                                     (short)0, acc, false, false);
    }

    const float scale = (MODE == 0) ? 0.125f : 1.0f;   // dh^-0.5 = 1/8
    const int col = n0 + laneM;
    const int h = col >> 6, d = col & 63;
    #pragma unroll
    for (int r = 0; r < 8; ++r) {
        int row = m0 + r + hi * 8;          // b*L + q
        int b = row >> 12, q = row & (L_SEQ - 1);
        int bh = b * N_HEAD + h;
        _Float16 hv = (_Float16)(acc[r] * scale);
        size_t idx;
        if (MODE == 0)      idx = (((size_t)bh * L_SEQ + q) * DH) + d;
        else if (MODE == 1) idx = (((size_t)bh * 32 + (d >> 1)) * L_SEQ + q) * 2 + (d & 1);
        else                idx = (((size_t)bh * (L_SEQ / 2) + (q >> 1)) * DH + d) * 2 + (q & 1);
        out[idx] = hv;
    }
}

// ---------------------------------------------------------------------------
// Fused flash attention, fixed-shift softmax:
//   logits are clamped to [-15,15] by the reference, so exp(x - 15) cannot
//   overflow and the softmax shift can be the constant 15 instead of the row
//   max. No running max, no alpha rescale, no cross-lane reductions: the row
//   sum is accumulated by an extra WMMA against an all-ones B fragment
//   (result replicated across all 16 columns -> shuffle-free epilogue).
// ---------------------------------------------------------------------------
__global__ __launch_bounds__(128)
void attn_kernel(const _Float16* __restrict__ Qh,   // [BH, L, 64]
                 const _Float16* __restrict__ Kt,   // [BH, 32, L, 2]
                 const _Float16* __restrict__ Vp,   // [BH, 2048, 64, 2]
                 const float*    __restrict__ mask, // [B, L, L]
                 float*          __restrict__ out)  // [B, L, 512]
{
    __shared__ _Float16 lds[4][16 * 34];            // per-wave padded 16x32 P tile

    const int lane  = threadIdx.x & 31;
    const int wave  = threadIdx.x >> 5;
    const int laneM = lane & 15;
    const int hi    = (lane >> 4) & 1;
    const int bh    = blockIdx.x;                   // 0..15
    const int b     = bh >> 3, h = bh & 7;
    const int q0    = (blockIdx.y * 4 + wave) * 16;

    // Q A-fragments for d-chunks [0,32) and [32,64): packed pairs are contiguous
    Frag16 aq[2];
    {
        const unsigned int* qrow =
            (const unsigned int*)(Qh + ((size_t)bh * L_SEQ + q0 + laneM) * DH);
        #pragma unroll
        for (int c = 0; c < 2; ++c)
            #pragma unroll
            for (int v = 0; v < 8; ++v) {
                int k = c * 32 + hi * 16 + 2 * v;
                aq[c].u[v] = qrow[k >> 1];
            }
    }

    // All-ones B fragment for the row-sum WMMA
    Frag16 ones;
    #pragma unroll
    for (int v = 0; v < 8; ++v) {
        ones.h[2 * v]     = (_Float16)1.0f;
        ones.h[2 * v + 1] = (_Float16)1.0f;
    }

    v8f o[4] = {};
    v8f osum = {};

    const float* mrow = mask + ((size_t)b * L_SEQ + q0) * L_SEQ;
    const unsigned int* ktb = (const unsigned int*)Kt + (size_t)bh * 32 * L_SEQ;
    const unsigned int* vpb = (const unsigned int*)Vp + (size_t)bh * (L_SEQ / 2) * DH;
    _Float16* myLds = lds[wave];

    for (int kv = 0; kv < L_SEQ; kv += 32) {
        // ---- prefetch kv block two ahead (K, V, mask) -> global_prefetch_b8
        if (kv + 64 < L_SEQ) {
            __builtin_prefetch(&ktb[(size_t)(hi * 8) * L_SEQ + kv + 64 + laneM], 0, 1);
            __builtin_prefetch(&vpb[(size_t)((kv + 64) >> 1) * DH + lane * 2], 0, 1);
            __builtin_prefetch(&mrow[(size_t)(hi * 8) * L_SEQ + kv + 64 + laneM], 0, 1);
        }

        // ---- S = Q * K^T : two 16x16 kv tiles, each accumulating 2 d-chunks
        v8f s[2];
        #pragma unroll
        for (int t = 0; t < 2; ++t) {
            Frag16 bk0, bk1;
            int kcol = kv + t * 16 + laneM;
            #pragma unroll
            for (int v = 0; v < 8; ++v) {
                int dp = hi * 8 + v;                 // d-pair index within chunk
                bk0.u[v] = ktb[(size_t)dp * L_SEQ + kcol];
                bk1.u[v] = ktb[(size_t)(dp + 16) * L_SEQ + kcol];
            }
            v8f z = {};
            z = __builtin_amdgcn_wmma_f32_16x16x32_f16(false, aq[0].h, false, bk0.h,
                                                       (short)0, z, false, false);
            z = __builtin_amdgcn_wmma_f32_16x16x32_f16(false, aq[1].h, false, bk1.h,
                                                       (short)0, z, false, false);
            s[t] = z;
        }

        // ---- P = exp(clamp(s)*m - 15) * m ; stage P tile to LDS (C -> A layout)
        #pragma unroll
        for (int t = 0; t < 2; ++t)
            #pragma unroll
            for (int r = 0; r < 8; ++r) {
                int q = r + hi * 8;
                int k = kv + t * 16 + laneM;
                float mk = mrow[(size_t)q * L_SEQ + k];
                float sv = fminf(15.0f, fmaxf(-15.0f, s[t][r])) * mk;
                float pv = __expf(sv - 15.0f) * mk;
                myLds[q * 34 + t * 16 + laneM] = (_Float16)pv;
            }

        __builtin_amdgcn_wave_barrier();   // order LDS stores before loads (in-wave)

        // ---- read P back in A-fragment layout (pairs contiguous in LDS row)
        Frag16 pa;
        #pragma unroll
        for (int v = 0; v < 8; ++v) {
            int k = hi * 16 + 2 * v;
            pa.u[v] = *(const unsigned int*)(myLds + laneM * 34 + k);
        }

        __builtin_amdgcn_wave_barrier();

        // ---- row-sum via matrix pipe: osum += P * ones
        osum = __builtin_amdgcn_wmma_f32_16x16x32_f16(false, pa.h, false, ones.h,
                                                      (short)0, osum, false, false);

        // ---- O += P * V over dh = 4 column tiles of 16
        #pragma unroll
        for (int c = 0; c < 4; ++c) {
            Frag16 bv;
            int d = c * 16 + laneM;
            #pragma unroll
            for (int v = 0; v < 8; ++v) {
                int kp = (kv >> 1) + hi * 8 + v;     // kv-pair row
                bv.u[v] = vpb[(size_t)kp * DH + d];
            }
            o[c] = __builtin_amdgcn_wmma_f32_16x16x32_f16(false, pa.h, false, bv.h,
                                                          (short)0, o[c], false, false);
        }
    }

    // ---- epilogue: normalize (rowsum replicated in every column) and write
    #pragma unroll
    for (int c = 0; c < 4; ++c)
        #pragma unroll
        for (int r = 0; r < 8; ++r) {
            int q = q0 + r + hi * 8;
            int dcol = h * DH + c * 16 + laneM;
            out[((size_t)b * L_SEQ + q) * D_MODEL + dcol] = o[c][r] / (osum[r] + 1e-6f);
        }
}

// ---------------------------------------------------------------------------
extern "C" void kernel_launch(void* const* d_in, const int* in_sizes, int n_in,
                              void* d_out, int out_size, void* d_ws, size_t ws_size,
                              hipStream_t stream) {
    const float* queries    = (const float*)d_in[0];
    const float* query_mask = (const float*)d_in[1];
    const float* keys       = (const float*)d_in[2];
    const float* values     = (const float*)d_in[3];
    const float* Wq         = (const float*)d_in[4];
    const float* Wk         = (const float*)d_in[5];
    const float* Wv         = (const float*)d_in[6];
    float* out = (float*)d_out;

    const size_t elems = (size_t)BATCH * N_HEAD * L_SEQ * DH;   // 4M f16 = 8MB each
    _Float16* Qh = (_Float16*)d_ws;
    _Float16* Kt = Qh + elems;
    _Float16* Vp = Kt + elems;

    dim3 pg(BATCH * L_SEQ / 16, D_MODEL / 64, 1);   // (512, 8)
    proj_kernel<0><<<pg, 128, 0, stream>>>(queries, Wq, Qh);
    proj_kernel<1><<<pg, 128, 0, stream>>>(keys,    Wk, Kt);
    proj_kernel<2><<<pg, 128, 0, stream>>>(values,  Wv, Vp);

    dim3 ag(BATCH * N_HEAD, L_SEQ / 64, 1);         // (16, 64)
    attn_kernel<<<ag, 128, 0, stream>>>(Qh, Kt, Vp, query_mask, out);
}
